// NonLocalBlock_6760278524120
// MI455X (gfx1250) — matrix operand order repaired and checked by reading, not compiled, not run on hardware
//
#include <hip/hip_runtime.h>
#include <hip/hip_bf16.h>

#define IN_CH 64
#define INTER 32
#define NB    2
#define NPOS  6400        // 80*80
#define QT    (NPOS/16)   // 400 query tiles per batch
#define NPART 4           // key-dimension split for occupancy
#define KSPAN (NPOS/NPART)

typedef __bf16        v16bf __attribute__((ext_vector_type(16)));
typedef float         v8f   __attribute__((ext_vector_type(8)));
typedef unsigned int  u32x4 __attribute__((ext_vector_type(4)));

union BV {
    u32x4 u[2];
    v16bf v;
};

// ---------------------------------------------------------------------------
// Kernel 1: channel projections.  theta/phi stored row-major [B][N][32] bf16,
// g stored transposed [B][32][N] bf16 (direct B-operand of the P*G WMMA).
// ---------------------------------------------------------------------------
__global__ void nlb_proj(const float* __restrict__ x,
                         const float* __restrict__ g_w,
                         const float* __restrict__ th_w,
                         const float* __restrict__ ph_w,
                         __bf16* __restrict__ thetaB,
                         __bf16* __restrict__ phiB,
                         __bf16* __restrict__ gT) {
    __shared__ float wg[INTER * IN_CH];
    __shared__ float wt[INTER * IN_CH];
    __shared__ float wp[INTER * IN_CH];
    const int t = threadIdx.x;
    for (int i = t; i < INTER * IN_CH; i += 128) {
        wg[i] = g_w[i];
        wt[i] = th_w[i];
        wp[i] = ph_w[i];
    }
    __syncthreads();

    const int gid = blockIdx.x * 128 + t;
    const int b = gid / NPOS;
    const int n = gid % NPOS;

    float xr[IN_CH];
    const float* xp = x + (size_t)b * IN_CH * NPOS + n;
#pragma unroll
    for (int c = 0; c < IN_CH; ++c) xr[c] = xp[(size_t)c * NPOS];

    __bf16* trow = thetaB + ((size_t)b * NPOS + n) * INTER;
    __bf16* prow = phiB   + ((size_t)b * NPOS + n) * INTER;
    __bf16* gcol = gT + (size_t)b * INTER * NPOS + n;

    for (int i = 0; i < INTER; ++i) {
        float at = 0.f, ap = 0.f, ag = 0.f;
#pragma unroll
        for (int c = 0; c < IN_CH; ++c) {
            const float xv = xr[c];
            at = fmaf(wt[i * IN_CH + c], xv, at);
            ap = fmaf(wp[i * IN_CH + c], xv, ap);
            ag = fmaf(wg[i * IN_CH + c], xv, ag);
        }
        trow[i] = (__bf16)at;
        prow[i] = (__bf16)ap;
        gcol[(size_t)i * NPOS] = (__bf16)ag;
    }
}

__global__ void nlb_wconv(const float* __restrict__ w, __bf16* __restrict__ wbf) {
    const int i = blockIdx.x * 256 + threadIdx.x;
    if (i < IN_CH * INTER) wbf[i] = (__bf16)w[i];
}

// ---------------------------------------------------------------------------
// Core flash-attention partial over keys [k0,k1): returns per-query partial
// max/sum and unnormalized Y accumulators.  One wave = one 16-query tile.
// ---------------------------------------------------------------------------
__device__ __forceinline__
void attn_partial(const __bf16* thetaB, const __bf16* phiB, const __bf16* gT,
                  int b, int q0, int k0, int k1, int lane, int col, int hi,
                  v8f& Y0, v8f& Y1, float& m_row, float& l_row) {
    BV Bt;
    {
        const char* p = (const char*)thetaB +
                        ((size_t)b * NPOS + q0 + col) * (INTER * 2) + hi * 16;
        Bt.u[0] = *(const u32x4*)p;
        Bt.u[1] = *(const u32x4*)(p + 32);
    }
    const char* phiRow0 = (const char*)phiB + (size_t)b * NPOS * (INTER * 2) +
                          (size_t)col * (INTER * 2) + hi * 16;
    const char* g0base = (const char*)gT +
                         ((size_t)b * INTER + col     ) * (NPOS * 2) + hi * 16;
    const char* g1base = (const char*)gT +
                         ((size_t)b * INTER + col + 16) * (NPOS * 2) + hi * 16;
    const v8f Zc = {0, 0, 0, 0, 0, 0, 0, 0};

    for (int kb = k0; kb < k1; kb += 32) {
        BV A0, A1;
        {
            const char* p = phiRow0 + (size_t)kb * (INTER * 2);
            A0.u[0] = *(const u32x4*)p;
            A0.u[1] = *(const u32x4*)(p + 32);
            p += 16 * (INTER * 2);
            A1.u[0] = *(const u32x4*)p;
            A1.u[1] = *(const u32x4*)(p + 32);
        }
        v8f S0 = __builtin_amdgcn_wmma_f32_16x16x32_bf16(
            false, A0.v, false, Bt.v, (short)0, Zc, false, false);
        v8f S1 = __builtin_amdgcn_wmma_f32_16x16x32_bf16(
            false, A1.v, false, Bt.v, (short)0, Zc, false, false);

        float mloc = S0[0];
#pragma unroll
        for (int r = 1; r < 8; ++r) mloc = fmaxf(mloc, S0[r]);
#pragma unroll
        for (int r = 0; r < 8; ++r) mloc = fmaxf(mloc, S1[r]);
        mloc = fmaxf(mloc, __shfl_xor(mloc, 16, 32));
        const float mnew  = fmaxf(m_row, mloc);
        const float scale = __expf(m_row - mnew);

        float p0[8], p1[8];
        float ssum = 0.f;
#pragma unroll
        for (int r = 0; r < 8; ++r) {
            p0[r] = __expf(S0[r] - mnew); ssum += p0[r];
            p1[r] = __expf(S1[r] - mnew); ssum += p1[r];
        }
        ssum += __shfl_xor(ssum, 16, 32);
        l_row = l_row * scale + ssum;
        m_row = mnew;

        v16bf Ap;
#pragma unroll
        for (int e = 0; e < 8; ++e) {
            Ap[e]     = (__bf16)p0[e];
            Ap[e + 8] = (__bf16)p1[e];
        }
#pragma unroll
        for (int r = 0; r < 8; ++r) {
            const float sc = __shfl(scale, r + hi * 8, 32);
            Y0[r] *= sc;
            Y1[r] *= sc;
        }
        BV G0, G1;
        {
            const char* p = g0base + (size_t)kb * 2;
            G0.u[0] = *(const u32x4*)p;
            G0.u[1] = *(const u32x4*)(p + 32);
            p = g1base + (size_t)kb * 2;
            G1.u[0] = *(const u32x4*)p;
            G1.u[1] = *(const u32x4*)(p + 32);
        }
        Y0 = __builtin_amdgcn_wmma_f32_16x16x32_bf16(
            false, Ap, false, G0.v, (short)0, Y0, false, false);
        Y1 = __builtin_amdgcn_wmma_f32_16x16x32_bf16(
            false, Ap, false, G1.v, (short)0, Y1, false, false);
    }
}

// Output projection + residual from a normalized Y tile staged in LDS.
__device__ __forceinline__
void project_out(const float* x, const __bf16* wbf, float* out,
                 __bf16* yb /*16x32 bf16 tile, this wave's LDS*/,
                 int b, int q0, int col, int hi) {
    BV Ay;
    {
        const char* p = (const char*)(yb + (size_t)col * INTER) + hi * 16;
        Ay.u[0] = *(const u32x4*)p;
        Ay.u[1] = *(const u32x4*)(p + 32);
    }
    const v8f Zc = {0, 0, 0, 0, 0, 0, 0, 0};
#pragma unroll
    for (int t4 = 0; t4 < 4; ++t4) {
        BV Bw;
        const char* p = (const char*)wbf +
                        (size_t)(t4 * 16 + col) * (INTER * 2) + hi * 16;
        Bw.u[0] = *(const u32x4*)p;
        Bw.u[1] = *(const u32x4*)(p + 32);
        v8f O = __builtin_amdgcn_wmma_f32_16x16x32_bf16(
            false, Ay.v, false, Bw.v, (short)0, Zc, false, false);
#pragma unroll
        for (int r = 0; r < 8; ++r) {
            const size_t c   = (size_t)(t4 * 16 + col);
            const size_t q   = (size_t)(q0 + r + hi * 8);
            const size_t idx = ((size_t)b * IN_CH + c) * NPOS + q;
            out[idx] = x[idx] + O[r];
        }
    }
}

// ---------------------------------------------------------------------------
// Kernel 2a (split-K): one wave = (partition, query-tile).  Blocks of 8 waves
// share a partition across adjacent tiles -> identical phi/g addresses hit
// the WGP cache.  Writes partial m, l, Y to workspace.
// ---------------------------------------------------------------------------
__global__ void __launch_bounds__(256)
nlb_attn_part(const __bf16* __restrict__ thetaB,
              const __bf16* __restrict__ phiB,
              const __bf16* __restrict__ gT,
              float* __restrict__ ypart,
              float* __restrict__ mpart,
              float* __restrict__ lpart) {
    const int lane = threadIdx.x & 31;
    const int wid  = threadIdx.x >> 5;
    const int work = blockIdx.x * 8 + wid;
    if (work >= NB * QT * NPART) return;          // wave-uniform
    const int part = work / (NB * QT);
    const int tile = work % (NB * QT);
    const int b   = tile / QT;
    const int q0  = (tile % QT) * 16;
    const int col = lane & 15;
    const int hi  = lane >> 4;

    v8f Y0 = {0, 0, 0, 0, 0, 0, 0, 0};
    v8f Y1 = {0, 0, 0, 0, 0, 0, 0, 0};
    float m_row = -3.0e38f;
    float l_row = 0.f;
    attn_partial(thetaB, phiB, gT, b, q0, part * KSPAN, (part + 1) * KSPAN,
                 lane, col, hi, Y0, Y1, m_row, l_row);

    const size_t slot = (size_t)tile * NPART + part;
    float* yp = ypart + slot * (16 * INTER);
#pragma unroll
    for (int r = 0; r < 8; ++r) {
        const int q = r + hi * 8;
        yp[q * INTER + col]      = Y0[r];
        yp[q * INTER + 16 + col] = Y1[r];
    }
    if (lane < 16) {                              // lane == q, dup at lane+16
        mpart[slot * 16 + lane] = m_row;
        lpart[slot * 16 + lane] = l_row;
    }
}

// ---------------------------------------------------------------------------
// Kernel 2b: merge NPART partials (log-sum-exp rescale), normalize, then
// WMMA output projection + residual.  One wave = one query tile.
// ---------------------------------------------------------------------------
__global__ void __launch_bounds__(256)
nlb_combine(const float* __restrict__ x,
            const float* __restrict__ ypart,
            const float* __restrict__ mpart,
            const float* __restrict__ lpart,
            const __bf16* __restrict__ wbf,
            float* __restrict__ out) {
    __shared__ __bf16 ylds[8 * 16 * INTER];
    const int lane = threadIdx.x & 31;
    const int wid  = threadIdx.x >> 5;
    const int tile = blockIdx.x * 8 + wid;
    if (tile >= NB * QT) return;
    const int b   = tile / QT;
    const int q0  = (tile % QT) * 16;
    const int col = lane & 15;
    const int hi  = lane >> 4;

    const size_t base = (size_t)tile * NPART;
    float m[NPART], l[NPART];
#pragma unroll
    for (int p = 0; p < NPART; ++p) {
        m[p] = mpart[(base + p) * 16 + col];
        l[p] = lpart[(base + p) * 16 + col];
    }
    float M = m[0];
#pragma unroll
    for (int p = 1; p < NPART; ++p) M = fmaxf(M, m[p]);
    float w[NPART], L = 0.f;
#pragma unroll
    for (int p = 0; p < NPART; ++p) {
        w[p] = __expf(m[p] - M);
        L += l[p] * w[p];
    }
    const float Linv = 1.f / L;

    // lane handles row q=col, inter slice i = hi*16 .. hi*16+15
    float acc[16];
#pragma unroll
    for (int e = 0; e < 16; ++e) acc[e] = 0.f;
#pragma unroll
    for (int p = 0; p < NPART; ++p) {
        const float* yp = ypart + (base + p) * (16 * INTER) +
                          (size_t)col * INTER + hi * 16;
#pragma unroll
        for (int e = 0; e < 16; ++e) acc[e] = fmaf(yp[e], w[p], acc[e]);
    }

    union { u32x4 u[2]; __bf16 h[16]; } st;
#pragma unroll
    for (int e = 0; e < 16; ++e) st.h[e] = (__bf16)(acc[e] * Linv);
    __bf16* yb = ylds + wid * (16 * INTER);
    u32x4* dst = (u32x4*)((char*)(yb + (size_t)col * INTER) + hi * 32);
    dst[0] = st.u[0];
    dst[1] = st.u[1];
    // Same wave wrote the full 16x32 tile; wave-internal DS is in-order.

    project_out(x, wbf, out, yb, b, q0, col, hi);
}

// ---------------------------------------------------------------------------
// Fallback single-pass fused kernel (used only if workspace is too small for
// split-K partials).  One wave = one query tile over all keys.
// ---------------------------------------------------------------------------
__global__ void __launch_bounds__(256)
nlb_attn_fused(const float* __restrict__ x,
               const __bf16* __restrict__ thetaB,
               const __bf16* __restrict__ phiB,
               const __bf16* __restrict__ gT,
               const __bf16* __restrict__ wbf,
               float* __restrict__ out) {
    __shared__ __bf16 ylds[8 * 16 * INTER];
    const int lane = threadIdx.x & 31;
    const int wid  = threadIdx.x >> 5;
    const int tile = blockIdx.x * 8 + wid;
    if (tile >= NB * QT) return;
    const int b   = tile / QT;
    const int q0  = (tile % QT) * 16;
    const int col = lane & 15;
    const int hi  = lane >> 4;

    v8f Y0 = {0, 0, 0, 0, 0, 0, 0, 0};
    v8f Y1 = {0, 0, 0, 0, 0, 0, 0, 0};
    float m_row = -3.0e38f;
    float l_row = 0.f;
    attn_partial(thetaB, phiB, gT, b, q0, 0, NPOS,
                 lane, col, hi, Y0, Y1, m_row, l_row);

    const float li = 1.0f / l_row;
    __bf16* yb = ylds + wid * (16 * INTER);
#pragma unroll
    for (int r = 0; r < 8; ++r) {
        const float lr = __shfl(li, r + hi * 8, 32);
        const int q = r + hi * 8;
        yb[q * INTER + col]      = (__bf16)(Y0[r] * lr);
        yb[q * INTER + 16 + col] = (__bf16)(Y1[r] * lr);
    }
    project_out(x, wbf, out, yb, b, q0, col, hi);
}

// ---------------------------------------------------------------------------
extern "C" void kernel_launch(void* const* d_in, const int* in_sizes, int n_in,
                              void* d_out, int out_size, void* d_ws, size_t ws_size,
                              hipStream_t stream) {
    (void)in_sizes; (void)n_in; (void)out_size;
    const float* x    = (const float*)d_in[0];
    const float* g_w  = (const float*)d_in[1];
    const float* th_w = (const float*)d_in[2];
    const float* ph_w = (const float*)d_in[3];
    const float* w_w  = (const float*)d_in[4];
    float* out = (float*)d_out;

    char* ws = (char*)d_ws;
    const size_t szP = (size_t)NB * NPOS * INTER * sizeof(__bf16); // 819200 B
    __bf16* thetaB = (__bf16*)(ws);
    __bf16* phiB   = (__bf16*)(ws + szP);
    __bf16* gTm    = (__bf16*)(ws + 2 * szP);
    __bf16* wbf    = (__bf16*)(ws + 3 * szP);
    const size_t baseBytes = 3 * szP + 4096;

    const size_t nslot   = (size_t)NB * QT * NPART;
    const size_t ypBytes = nslot * 16 * INTER * sizeof(float);   // 6.55 MB
    const size_t mlBytes = nslot * 16 * sizeof(float);           // 0.2 MB each
    float* ypart = (float*)(ws + baseBytes);
    float* mpart = (float*)(ws + baseBytes + ypBytes);
    float* lpart = (float*)(ws + baseBytes + ypBytes + mlBytes);
    const bool splitk = ws_size >= baseBytes + ypBytes + 2 * mlBytes;

    nlb_proj<<<(NB * NPOS) / 128, 128, 0, stream>>>(x, g_w, th_w, ph_w,
                                                    thetaB, phiB, gTm);
    nlb_wconv<<<(IN_CH * INTER + 255) / 256, 256, 0, stream>>>(w_w, wbf);

    if (splitk) {
        nlb_attn_part<<<(NB * QT * NPART) / 8, 256, 0, stream>>>(
            thetaB, phiB, gTm, ypart, mpart, lpart);
        nlb_combine<<<(NB * QT) / 8, 256, 0, stream>>>(
            x, ypart, mpart, lpart, wbf, out);
    } else {
        nlb_attn_fused<<<(NB * QT) / 8, 256, 0, stream>>>(
            x, thetaB, phiB, gTm, wbf, out);
    }
}
